// HeatDiffusionICGRNN_28527172780597
// MI455X (gfx1250) — matrix operand order repaired and checked by reading, not compiled, not run on hardware
//
#include <hip/hip_runtime.h>

// ---------------------------------------------------------------------------
// HeatDiffusionICGRNN on gfx1250 (MI455X).
// Edge message ICNN (95% of FLOPs) runs on v_wmma_f32_16x16x32_f16 with f32
// accumulate; weights pre-softplus'd and packed into B-fragment layout.
// Per-node ICNNs / layernorms use wave-per-node VALU with __shfl broadcasts.
// ---------------------------------------------------------------------------

#define D_IN   8
#define HIDN   64
#define H1     32
#define H2     32
#define SCALEF 0.1f
#define LN_EPS 1e-5f

typedef __attribute__((ext_vector_type(16))) _Float16 v16h;
typedef __attribute__((ext_vector_type(8)))  float    v8f;

union AFrag { v16h v; unsigned u[8]; };

#define WMMA_F16(A, B, C) \
  __builtin_amdgcn_wmma_f32_16x16x32_f16(false, (A), false, (B), (short)0, (C), false, false)

// weight-pack fragment slots (each slot = 32 lanes * 16 halfs = 512 halfs)
#define SLOT_W0(kb, nb)  ((kb) * 2 + (nb))
#define SLOT_WZ(nb)      (4 + (nb))
#define SLOT_WX(kb, nb)  (6 + (kb) * 2 + (nb))
#define SLOT_WZO(nb)     (10 + (nb))
#define SLOT_WXO(kb, nb) (14 + (kb) * 4 + (nb))
#define N_SLOTS 22

__device__ __forceinline__ float softplusf(float v) {
  return (v > 20.f) ? v : log1pf(expf(v));
}

__device__ __forceinline__ float wred(float v) {
#pragma unroll
  for (int o = 16; o > 0; o >>= 1) v += __shfl_xor(v, o, 32);
  return v;
}

// ---------------------------------------------------------------------------
__global__ __launch_bounds__(256) void zero_kernel(float* __restrict__ p, long long n) {
  long long i = (long long)blockIdx.x * blockDim.x + threadIdx.x;
  long long stride = (long long)gridDim.x * blockDim.x;
  for (; i < n; i += stride) p[i] = 0.f;
}

// ---------------------------------------------------------------------------
// Softplus copies of all z-path weights + f16 B-fragment packing of p_mp.
__global__ __launch_bounds__(256) void prep_kernel(
    const float* __restrict__ ihWz, const float* __restrict__ ihWzo,
    const float* __restrict__ hhWz, const float* __restrict__ hhWzo,
    const float* __restrict__ tWz,  const float* __restrict__ tWzo,
    const float* __restrict__ cWz,  const float* __restrict__ cWzo,
    const float* __restrict__ mpW0, const float* __restrict__ mpWz,
    const float* __restrict__ mpWx, const float* __restrict__ mpWzo,
    const float* __restrict__ mpWxo,
    float* __restrict__ sp, _Float16* __restrict__ wpack) {
  int t = blockIdx.x * blockDim.x + threadIdx.x;
  if (t < 8256) {
    const float* src; float* dst; int off;
    if      (t < 1024) { src = ihWz;  dst = sp;        off = t; }
    else if (t < 3072) { src = ihWzo; dst = sp + 1024; off = t - 1024; }
    else if (t < 4096) { src = hhWz;  dst = sp + 3072; off = t - 3072; }
    else if (t < 6144) { src = hhWzo; dst = sp + 4096; off = t - 4096; }
    else if (t < 7168) { src = tWz;   dst = sp + 6144; off = t - 6144; }
    else if (t < 7200) { src = tWzo;  dst = sp + 7168; off = t - 7168; }
    else if (t < 8224) { src = cWz;   dst = sp + 7200; off = t - 7200; }
    else               { src = cWzo;  dst = sp + 8224; off = t - 8224; }
    dst[off] = softplusf(src[off]);
  }
  int p = t - 8256;
  if (p >= 0 && p < N_SLOTS * 512) {
    int slot = p >> 9;
    int r = p & 511;
    int l = r >> 4, j = r & 15;
    const float* src; int outD, kb, nb; bool dosp = false;
    if (slot < 4)       { src = mpW0;  outD = 32; kb = slot >> 1; nb = slot & 1; }
    else if (slot < 6)  { src = mpWz;  outD = 32; kb = 0; nb = slot - 4; dosp = true; }
    else if (slot < 10) { int q = slot - 6;  src = mpWx;  outD = 32; kb = q >> 1; nb = q & 1; }
    else if (slot < 14) { src = mpWzo; outD = 64; kb = 0; nb = slot - 10; dosp = true; }
    else                { int q = slot - 14; src = mpWxo; outD = 64; kb = q >> 2; nb = q & 3; }
    // B-fragment (16-bit, 32x16): lane l holds col n=l%16, K = (j<8 ? j : j+8) + (l>=16 ? 8 : 0)
    int K = 32 * kb + ((j < 8) ? j : (j + 8)) + ((l >= 16) ? 8 : 0);
    int n = 16 * nb + (l & 15);
    float v = src[K * outD + n];
    if (dosp) v = softplusf(v);
    wpack[p] = (_Float16)v;
  }
}

// ---------------------------------------------------------------------------
__global__ __launch_bounds__(256) void cnt_kernel(const int* __restrict__ eidx,
                                                  float* __restrict__ cnt,
                                                  int E, int N, int total) {
  int t = blockIdx.x * blockDim.x + threadIdx.x;
  if (t < total) {
    int b = t / E, e = t - b * E;
    atomicAdd(&cnt[eidx[E + e] + b * N], 1.f);
  }
}

// ---------------------------------------------------------------------------
// Input ICNN (d_in=8) -> input_transform = SCALE * icnn(x).  Wave per node.
__global__ __launch_bounds__(256) void ih_kernel(
    const float* __restrict__ x,
    const float* __restrict__ W0, const float* __restrict__ b0,
    const float* __restrict__ spWz, const float* __restrict__ Wx,
    const float* __restrict__ b1, const float* __restrict__ spWzo,
    const float* __restrict__ Wxo, const float* __restrict__ bo,
    float* __restrict__ it, int BN) {
  int lane = threadIdx.x & 31;
  int gw = (blockIdx.x * blockDim.x + threadIdx.x) >> 5;
  int nw = (gridDim.x * blockDim.x) >> 5;
  for (int node = gw; node < BN; node += nw) {
    float xv = (lane < D_IN) ? x[(size_t)node * D_IN + lane] : 0.f;
    float z = b0[lane];
#pragma unroll
    for (int k = 0; k < D_IN; ++k)
      z = fmaf(__shfl(xv, k, 32), W0[k * 32 + lane], z);
    z = fmaxf(z, 0.f);
    float z1 = b1[lane];
#pragma unroll
    for (int k = 0; k < 32; ++k)
      z1 = fmaf(__shfl(z, k, 32), spWz[k * 32 + lane], z1);
#pragma unroll
    for (int k = 0; k < D_IN; ++k)
      z1 = fmaf(__shfl(xv, k, 32), Wx[k * 32 + lane], z1);
    z1 = fmaxf(z1, 0.f);
    float o0 = bo[lane], o1 = bo[lane + 32];
#pragma unroll
    for (int k = 0; k < 32; ++k) {
      float zv = __shfl(z1, k, 32);
      o0 = fmaf(zv, spWzo[k * 64 + lane], o0);
      o1 = fmaf(zv, spWzo[k * 64 + lane + 32], o1);
    }
#pragma unroll
    for (int k = 0; k < D_IN; ++k) {
      float xk = __shfl(xv, k, 32);
      o0 = fmaf(xk, Wxo[k * 64 + lane], o0);
      o1 = fmaf(xk, Wxo[k * 64 + lane + 32], o1);
    }
    size_t base = (size_t)node * HIDN;
    it[base + lane] = SCALEF * o0;
    it[base + lane + 32] = SCALEF * o1;
  }
}

// ---------------------------------------------------------------------------
// h_tilde = relu(IT + SCALE * icnn_hh(h)), stored as f16.  Wave per node.
__global__ __launch_bounds__(256) void hh_kernel(
    const float* __restrict__ h, const float* __restrict__ it,
    const float* __restrict__ W0, const float* __restrict__ b0,
    const float* __restrict__ spWz, const float* __restrict__ Wx,
    const float* __restrict__ b1, const float* __restrict__ spWzo,
    const float* __restrict__ Wxo, const float* __restrict__ bo,
    _Float16* __restrict__ ht16, int BN) {
  int lane = threadIdx.x & 31;
  int gw = (blockIdx.x * blockDim.x + threadIdx.x) >> 5;
  int nw = (gridDim.x * blockDim.x) >> 5;
  for (int node = gw; node < BN; node += nw) {
    size_t base = (size_t)node * HIDN;
    float x0 = h[base + lane], x1 = h[base + lane + 32];
    float z = b0[lane];
#pragma unroll
    for (int k = 0; k < 64; ++k)
      z = fmaf(__shfl(k < 32 ? x0 : x1, k & 31, 32), W0[k * 32 + lane], z);
    z = fmaxf(z, 0.f);
    float z1 = b1[lane];
#pragma unroll
    for (int k = 0; k < 32; ++k)
      z1 = fmaf(__shfl(z, k, 32), spWz[k * 32 + lane], z1);
#pragma unroll
    for (int k = 0; k < 64; ++k)
      z1 = fmaf(__shfl(k < 32 ? x0 : x1, k & 31, 32), Wx[k * 32 + lane], z1);
    z1 = fmaxf(z1, 0.f);
    float o0 = bo[lane], o1 = bo[lane + 32];
#pragma unroll
    for (int k = 0; k < 32; ++k) {
      float zv = __shfl(z1, k, 32);
      o0 = fmaf(zv, spWzo[k * 64 + lane], o0);
      o1 = fmaf(zv, spWzo[k * 64 + lane + 32], o1);
    }
#pragma unroll
    for (int k = 0; k < 64; ++k) {
      float xv = __shfl(k < 32 ? x0 : x1, k & 31, 32);
      o0 = fmaf(xv, Wxo[k * 64 + lane], o0);
      o1 = fmaf(xv, Wxo[k * 64 + lane + 32], o1);
    }
    float t0 = fmaxf(it[base + lane] + SCALEF * o0, 0.f);
    float t1 = fmaxf(it[base + lane + 32] + SCALEF * o1, 0.f);
    ht16[base + lane] = (_Float16)t0;
    ht16[base + lane + 32] = (_Float16)t1;
  }
}

// ---------------------------------------------------------------------------
// WMMA edge-message ICNN: per wave, 16 edges per tile.
// A-frags gathered from ht16, B-frags from packed weights, C in f32, scatter
// via global f32 atomics into h_raw.
__global__ __launch_bounds__(128) void mp_kernel(
    const _Float16* __restrict__ ht16, const int* __restrict__ eidx,
    const float* __restrict__ mb0, const float* __restrict__ mb1,
    const float* __restrict__ mbo, const _Float16* __restrict__ wpack,
    float* __restrict__ hraw, int E, int N, int ntiles) {
  __shared__ _Float16 stage[4][16 * 32];
  __shared__ int dsts[4][16];
  const int lane = threadIdx.x & 31;
  const int wv = threadIdx.x >> 5;
  const int l16 = lane & 15;
  const int hi8 = (lane >> 4) << 3;  // 0 or 8
  int gw = blockIdx.x * 4 + wv;
  int nw = gridDim.x * 4;

  // loop-invariant biases
  float bz0_0 = mb0[l16], bz0_1 = mb0[l16 + 16];
  float bz1_0 = mb1[l16], bz1_1 = mb1[l16 + 16];
  float bo_[4] = {mbo[l16], mbo[l16 + 16], mbo[l16 + 32], mbo[l16 + 48]};

  for (int tile = gw; tile < ntiles; tile += nw) {
    int eid = tile * 16 + l16;
    int b = eid / E;
    int e = eid - b * E;
    int srcN = eidx[e] + b * N;
    if (lane < 16) dsts[wv][l16] = eidx[E + e] + b * N;

    // Gather A fragments (16-bit 16x32 layout): lane l -> row l%16,
    // VGPR v holds K pair {2v,2v+1} (+8 for lanes>=16; v>=4 -> +16).
    const _Float16* row = ht16 + (size_t)srcN * HIDN;
    AFrag XA0, XA1;
#pragma unroll
    for (int v = 0; v < 8; ++v) {
      int k0 = ((v < 4) ? (v << 1) : (16 + ((v - 4) << 1))) + hi8;
      XA0.u[v] = *(const unsigned*)(row + k0);
      XA1.u[v] = *(const unsigned*)(row + 32 + k0);
    }

    // ---- layer 1: z0 = relu(X @ W0 + b0)   [16x32]
#pragma unroll
    for (int nb = 0; nb < 2; ++nb) {
      v8f acc = {};
      acc = WMMA_F16(XA0.v, *(const v16h*)(wpack + (SLOT_W0(0, nb) << 9) + (lane << 4)), acc);
      acc = WMMA_F16(XA1.v, *(const v16h*)(wpack + (SLOT_W0(1, nb) << 9) + (lane << 4)), acc);
      float bb = nb ? bz0_1 : bz0_0;
#pragma unroll
      for (int r = 0; r < 8; ++r) {
        float z = fmaxf(acc[r] + bb, 0.f);
        stage[wv][(r + hi8) * 32 + l16 + (nb << 4)] = (_Float16)z;
      }
    }
    asm volatile("s_wait_dscnt 0" ::: "memory");
    AFrag Z0A;
#pragma unroll
    for (int v = 0; v < 8; ++v) {
      int k0 = ((v < 4) ? (v << 1) : (16 + ((v - 4) << 1))) + hi8;
      Z0A.u[v] = *(const unsigned*)&stage[wv][l16 * 32 + k0];
    }
    asm volatile("" ::: "memory");

    // ---- layer 2: z1 = relu(z0 @ spWz + X @ Wx + b)   [16x32]
#pragma unroll
    for (int nb = 0; nb < 2; ++nb) {
      v8f acc = {};
      acc = WMMA_F16(Z0A.v, *(const v16h*)(wpack + (SLOT_WZ(nb) << 9) + (lane << 4)), acc);
      acc = WMMA_F16(XA0.v, *(const v16h*)(wpack + (SLOT_WX(0, nb) << 9) + (lane << 4)), acc);
      acc = WMMA_F16(XA1.v, *(const v16h*)(wpack + (SLOT_WX(1, nb) << 9) + (lane << 4)), acc);
      float bb = nb ? bz1_1 : bz1_0;
#pragma unroll
      for (int r = 0; r < 8; ++r) {
        float z = fmaxf(acc[r] + bb, 0.f);
        stage[wv][(r + hi8) * 32 + l16 + (nb << 4)] = (_Float16)z;
      }
    }
    asm volatile("s_wait_dscnt 0" ::: "memory");
    AFrag Z1A;
#pragma unroll
    for (int v = 0; v < 8; ++v) {
      int k0 = ((v < 4) ? (v << 1) : (16 + ((v - 4) << 1))) + hi8;
      Z1A.u[v] = *(const unsigned*)&stage[wv][l16 * 32 + k0];
    }
    int d8[8];
#pragma unroll
    for (int r = 0; r < 8; ++r) d8[r] = dsts[wv][r + hi8];

    // ---- layer 3: out = z1 @ spWzo + X @ Wxo + bo   [16x64] -> atomic scatter
#pragma unroll
    for (int nb = 0; nb < 4; ++nb) {
      v8f acc = {};
      acc = WMMA_F16(Z1A.v, *(const v16h*)(wpack + (SLOT_WZO(nb) << 9) + (lane << 4)), acc);
      acc = WMMA_F16(XA0.v, *(const v16h*)(wpack + (SLOT_WXO(0, nb) << 9) + (lane << 4)), acc);
      acc = WMMA_F16(XA1.v, *(const v16h*)(wpack + (SLOT_WXO(1, nb) << 9) + (lane << 4)), acc);
      float bb = bo_[nb];
      int ncol = l16 + (nb << 4);
#pragma unroll
      for (int r = 0; r < 8; ++r)
        atomicAdd(&hraw[(size_t)d8[r] * HIDN + ncol], acc[r] + bb);
    }
  }
}

// ---------------------------------------------------------------------------
// Scalar-output ICNN over 64-dim input held 2 elems/lane.
__device__ __forceinline__ float icnn1(
    float x0, float x1, int lane,
    const float* __restrict__ W0, const float* __restrict__ b0,
    const float* __restrict__ spWz, const float* __restrict__ Wx,
    const float* __restrict__ b1, const float* __restrict__ spWzo,
    const float* __restrict__ Wxo, const float* __restrict__ bo) {
  float z = b0[lane];
#pragma unroll
  for (int k = 0; k < 64; ++k)
    z = fmaf(__shfl(k < 32 ? x0 : x1, k & 31, 32), W0[k * 32 + lane], z);
  z = fmaxf(z, 0.f);
  float z1 = b1[lane];
#pragma unroll
  for (int k = 0; k < 32; ++k)
    z1 = fmaf(__shfl(z, k, 32), spWz[k * 32 + lane], z1);
#pragma unroll
  for (int k = 0; k < 64; ++k)
    z1 = fmaf(__shfl(k < 32 ? x0 : x1, k & 31, 32), Wx[k * 32 + lane], z1);
  z1 = fmaxf(z1, 0.f);
  float part = z1 * spWzo[lane] + x0 * Wxo[lane] + x1 * Wxo[lane + 32];
  return wred(part) + bo[0];
}

// ---------------------------------------------------------------------------
// h_new = LN(h_raw/cnt); temp/ctrl heads; re-zero h_raw.  Wave per node.
__global__ __launch_bounds__(256) void ln_out_kernel(
    float* __restrict__ hraw, const float* __restrict__ cnt,
    float* __restrict__ h,
    const float* __restrict__ lnhg, const float* __restrict__ lnhb,
    const float* __restrict__ lntg, const float* __restrict__ lntb,
    const float* __restrict__ lncg, const float* __restrict__ lncb,
    const float* __restrict__ tW0, const float* __restrict__ tb0,
    const float* __restrict__ tspWz, const float* __restrict__ tWx,
    const float* __restrict__ tb1, const float* __restrict__ tspWzo,
    const float* __restrict__ tWxo, const float* __restrict__ tbo,
    const float* __restrict__ cW0, const float* __restrict__ cb0,
    const float* __restrict__ cspWz, const float* __restrict__ cWx,
    const float* __restrict__ cb1, const float* __restrict__ cspWzo,
    const float* __restrict__ cWxo, const float* __restrict__ cbo,
    float* __restrict__ out_t, float* __restrict__ out_c,
    float* __restrict__ out_h, int BN, int steps, int s, int writeH) {
  int lane = threadIdx.x & 31;
  int gw = (blockIdx.x * blockDim.x + threadIdx.x) >> 5;
  int nw = (gridDim.x * blockDim.x) >> 5;
  for (int node = gw; node < BN; node += nw) {
    size_t base = (size_t)node * HIDN;
    float c = fmaxf(cnt[node], 1.f);
    float a0 = hraw[base + lane] / c;
    float a1 = hraw[base + lane + 32] / c;
    hraw[base + lane] = 0.f;
    hraw[base + lane + 32] = 0.f;
    float m = wred(a0 + a1) * (1.f / 64.f);
    float vv = wred(a0 * a0 + a1 * a1) * (1.f / 64.f) - m * m;
    float inv = rsqrtf(vv + LN_EPS);
    float hn0 = (a0 - m) * inv * lnhg[lane] + lnhb[lane];
    float hn1 = (a1 - m) * inv * lnhg[lane + 32] + lnhb[lane + 32];
    h[base + lane] = hn0;
    h[base + lane + 32] = hn1;
    if (writeH) { out_h[base + lane] = hn0; out_h[base + lane + 32] = hn1; }
    float m2 = wred(hn0 + hn1) * (1.f / 64.f);
    float v2 = wred(hn0 * hn0 + hn1 * hn1) * (1.f / 64.f) - m2 * m2;
    float inv2 = rsqrtf(v2 + LN_EPS);
    float xt0 = (hn0 - m2) * inv2 * lntg[lane] + lntb[lane];
    float xt1 = (hn1 - m2) * inv2 * lntg[lane + 32] + lntb[lane + 32];
    float tval = icnn1(xt0, xt1, lane, tW0, tb0, tspWz, tWx, tb1, tspWzo, tWxo, tbo);
    float xc0 = (hn0 - m2) * inv2 * lncg[lane] + lncb[lane];
    float xc1 = (hn1 - m2) * inv2 * lncg[lane + 32] + lncb[lane + 32];
    float cval = icnn1(xc0, xc1, lane, cW0, cb0, cspWz, cWx, cb1, cspWzo, cWxo, cbo);
    if (lane == 0) {
      out_t[(size_t)node * steps + s] = fminf(fmaxf(tval, 0.f), 100.f);
      out_c[(size_t)node * steps + s] = fminf(fmaxf(cval, -50.f), 50.f);
    }
  }
}

// ---------------------------------------------------------------------------
extern "C" void kernel_launch(void* const* d_in, const int* in_sizes, int n_in,
                              void* d_out, int out_size, void* d_ws, size_t ws_size,
                              hipStream_t stream) {
  (void)n_in; (void)ws_size;
  const float* x    = (const float*)d_in[0];
  const int*   eidx = (const int*)d_in[1];
  // p_ih (d_in=8): W0,b0,Wz,Wx,b,Wzo,Wxo,bo
  const float *ihW0 = (const float*)d_in[2],  *ihb0 = (const float*)d_in[3];
  const float *ihWz = (const float*)d_in[4],  *ihWx = (const float*)d_in[5];
  const float *ihb  = (const float*)d_in[6],  *ihWzo = (const float*)d_in[7];
  const float *ihWxo = (const float*)d_in[8], *ihbo = (const float*)d_in[9];
  // p_hh
  const float *hhW0 = (const float*)d_in[10], *hhb0 = (const float*)d_in[11];
  const float *hhWz = (const float*)d_in[12], *hhWx = (const float*)d_in[13];
  const float *hhb  = (const float*)d_in[14], *hhWzo = (const float*)d_in[15];
  const float *hhWxo = (const float*)d_in[16], *hhbo = (const float*)d_in[17];
  // p_mp
  const float *mpW0 = (const float*)d_in[18], *mpb0 = (const float*)d_in[19];
  const float *mpWz = (const float*)d_in[20], *mpWx = (const float*)d_in[21];
  const float *mpb  = (const float*)d_in[22], *mpWzo = (const float*)d_in[23];
  const float *mpWxo = (const float*)d_in[24], *mpbo = (const float*)d_in[25];
  // p_temp
  const float *tW0 = (const float*)d_in[26], *tb0 = (const float*)d_in[27];
  const float *tWz = (const float*)d_in[28], *tWx = (const float*)d_in[29];
  const float *tb  = (const float*)d_in[30], *tWzo = (const float*)d_in[31];
  const float *tWxo = (const float*)d_in[32], *tbo = (const float*)d_in[33];
  // p_ctrl
  const float *cW0 = (const float*)d_in[34], *cb0 = (const float*)d_in[35];
  const float *cWz = (const float*)d_in[36], *cWx = (const float*)d_in[37];
  const float *cb  = (const float*)d_in[38], *cWzo = (const float*)d_in[39];
  const float *cWxo = (const float*)d_in[40], *cbo = (const float*)d_in[41];
  const float *lnhg = (const float*)d_in[42], *lnhb = (const float*)d_in[43];
  const float *lntg = (const float*)d_in[44], *lntb = (const float*)d_in[45];
  const float *lncg = (const float*)d_in[46], *lncb = (const float*)d_in[47];

  const int BN = in_sizes[0] / D_IN;        // B*N = 80000
  const int E  = in_sizes[1] / 2;           // 320000
  const int B  = 4;                         // per reference setup
  const int N  = BN / B;
  const int steps = (out_size - BN * HIDN) / (2 * BN);  // 20

  // workspace layout (floats): H | HRAW | CNT | IT | SP ; then WPACK/HT16 (f16)
  float* ws_f = (float*)d_ws;
  float* H    = ws_f;
  float* HRAW = H + (size_t)BN * HIDN;
  float* CNT  = HRAW + (size_t)BN * HIDN;
  float* IT   = CNT + BN;
  float* SP   = IT + (size_t)BN * HIDN;
  size_t fcount = (size_t)3 * BN * HIDN + BN + 8256;
  size_t wb = ((fcount * 4 + 31) / 32) * 32;
  _Float16* WPACK = (_Float16*)((char*)d_ws + wb);
  _Float16* HT16  = WPACK + N_SLOTS * 512;

  float* out_t = (float*)d_out;
  float* out_c = out_t + (size_t)BN * steps;
  float* out_h = out_t + (size_t)2 * BN * steps;

  long long zn = (long long)2 * BN * HIDN + BN;
  zero_kernel<<<2048, 256, 0, stream>>>(ws_f, zn);
  prep_kernel<<<(8256 + N_SLOTS * 512 + 255) / 256, 256, 0, stream>>>(
      ihWz, ihWzo, hhWz, hhWzo, tWz, tWzo, cWz, cWzo,
      mpW0, mpWz, mpWx, mpWzo, mpWxo, SP, WPACK);
  cnt_kernel<<<(B * E + 255) / 256, 256, 0, stream>>>(eidx, CNT, E, N, B * E);
  ih_kernel<<<640, 256, 0, stream>>>(x, ihW0, ihb0, SP + 0, ihWx, ihb,
                                     SP + 1024, ihWxo, ihbo, IT, BN);
  const int ntiles = (B * E) / 16;
  for (int s = 0; s < steps; ++s) {
    hh_kernel<<<640, 256, 0, stream>>>(H, IT, hhW0, hhb0, SP + 3072, hhWx, hhb,
                                       SP + 4096, hhWxo, hhbo, HT16, BN);
    mp_kernel<<<512, 128, 0, stream>>>(HT16, eidx, mpb0, mpb, mpbo, WPACK,
                                       HRAW, E, N, ntiles);
    ln_out_kernel<<<640, 256, 0, stream>>>(
        HRAW, CNT, H, lnhg, lnhb, lntg, lntb, lncg, lncb,
        tW0, tb0, SP + 6144, tWx, tb, SP + 7168, tWxo, tbo,
        cW0, cb0, SP + 7200, cWx, cb, SP + 8224, cWxo, cbo,
        out_t, out_c, out_h, BN, steps, s, (s == steps - 1) ? 1 : 0);
  }
}
